// DynamicGraphConv_11012296147596
// MI455X (gfx1250) — matrix operand order repaired and checked by reading, not compiled, not run on hardware
//
#include <hip/hip_runtime.h>
#include <hip/hip_bf16.h>
#include <math.h>

// ---------------------------------------------------------------------------
// DGCNN EdgeConv for MI455X (gfx1250, wave32, WMMA + TDM gather)
// ---------------------------------------------------------------------------

typedef _Float16 v16h __attribute__((ext_vector_type(16)));
typedef _Float16 h8   __attribute__((ext_vector_type(8)));
typedef float    v8f  __attribute__((ext_vector_type(8)));
typedef unsigned int u32x4 __attribute__((ext_vector_type(4)));
typedef unsigned int u32x8 __attribute__((ext_vector_type(8)));

#define BB    8
#define NN    4096
#define CC    64        // C_in == C_out
#define KNB   16        // neighbors
#define TWOC  128       // 2*C_in
#define NPTS  4         // points per edgeconv block (1 wave each)
#define ESTR  136       // padded LDS edge row stride in halves (272B, 16B aligned)
#define CNT_INV (1.0f / 524288.0f)   // 1/(B*N*K)

// -------------------- workspace layout (bytes) -----------------------------
#define WS_SUM   0u                    // 128 floats: [0..63]=sum, [64..127]=sumsq
#define WS_WH    512u                  // 64*128 _Float16 = 16384
#define WS_IDX   16896u                // 8*4096*16 u16   = 1048576
#define WS_HMAX  1065472u              // 8*4096*64 float = 8388608
#define WS_HMIN  9454080u              // 8*4096*64 float = 8388608
#define WS_PART  17842688u             // 8192*128 float  = 4194304
// total ~22.0 MB

// ---------------------------------------------------------------------------
// 1) Convert W (64 x 128 f32, row-major) to f16 in global (L2-resident)
// ---------------------------------------------------------------------------
__global__ void k_wconv(const float* __restrict__ W, _Float16* __restrict__ Wh) {
    int i = blockIdx.x * 256 + threadIdx.x;
    if (i < CC * TWOC) Wh[i] = (_Float16)W[i];
}

// ---------------------------------------------------------------------------
// 2) kNN: one thread per query point; 512-neighbor LDS tiles; u16 indices.
//    metric v = 2*dot(q,p) - |p|^2  (monotone-equivalent to reference pairwise)
// ---------------------------------------------------------------------------
__global__ __launch_bounds__(256) void k_knn(const float* __restrict__ xyz,
                                             unsigned short* __restrict__ idx16) {
    constexpr int TILE = 512;
    __shared__ float4 tile[TILE];

    const int b = blockIdx.x >> 4;                   // 16 blocks per batch
    const int n = ((blockIdx.x & 15) << 8) + threadIdx.x;
    const float* base = xyz + (size_t)b * NN * 3;

    const float qx = base[n * 3 + 0];
    const float qy = base[n * 3 + 1];
    const float qz = base[n * 3 + 2];

    float val[KNB];
    int   ind[KNB];
#pragma unroll
    for (int j = 0; j < KNB; ++j) { val[j] = -INFINITY; ind[j] = 0; }
    float vmin = -INFINITY;
    int   minpos = 0;

    for (int t0 = 0; t0 < NN; t0 += TILE) {
        __syncthreads();
        for (int i = threadIdx.x; i < TILE; i += 256) {
            const int m = t0 + i;
            const float px = base[m * 3 + 0];
            const float py = base[m * 3 + 1];
            const float pz = base[m * 3 + 2];
            tile[i] = make_float4(px, py, pz, px * px + py * py + pz * pz);
        }
        __syncthreads();
        for (int i = 0; i < TILE; ++i) {
            const float4 p = tile[i];
            const float v = 2.0f * (qx * p.x + qy * p.y + qz * p.z) - p.w;
            if (v > vmin) {
                const int m = t0 + i;
#pragma unroll
                for (int j = 0; j < KNB; ++j)
                    if (j == minpos) { val[j] = v; ind[j] = m; }
                vmin = val[0]; minpos = 0;
#pragma unroll
                for (int j = 1; j < KNB; ++j)
                    if (val[j] < vmin) { vmin = val[j]; minpos = j; }
            }
        }
    }
    const size_t gpt = (size_t)b * NN + n;
#pragma unroll
    for (int j = 0; j < KNB; ++j)
        idx16[gpt * KNB + j] = (unsigned short)ind[j];
}

// ---------------------------------------------------------------------------
// TDM issue helper: tensor_load_to_lds with 4 SGPR descriptor groups
// ---------------------------------------------------------------------------
__device__ __forceinline__ void tdm_load(u32x4 g0, u32x8 g1, u32x4 g2, u32x4 g3) {
    asm volatile("tensor_load_to_lds %0, %1, %2, %3"
                 :: "s"(g0), "s"(g1), "s"(g2), "s"(g3)
                 : "memory");
}

// ---------------------------------------------------------------------------
// 3) Fused TDM-gather + WMMA GEMM + per-point max/min + per-channel stats.
//    Block = 128 threads = 4 waves = 4 points. Deterministic reductions.
// ---------------------------------------------------------------------------
__global__ __launch_bounds__(128) void k_edgeconv(
    const float* __restrict__ x, const float* __restrict__ bias,
    const _Float16* __restrict__ Wh, const unsigned short* __restrict__ idx16,
    float* __restrict__ hmax, float* __restrict__ hmin,
    float* __restrict__ partial) {

    __shared__ float    xrows[NPTS * KNB * CC];        // 16384 B (TDM gather dst)
    __shared__ _Float16 wlds[CC * TWOC];               // 16384 B (TDM W dst)
    __shared__ _Float16 edge[NPTS * KNB * ESTR];       // 17408 B
    __shared__ float    wsum[NPTS * CC];               // 1024 B
    __shared__ float    wsumsq[NPTS * CC];             // 1024 B

    const int tid    = threadIdx.x;
    const int lane   = tid & 31;
    const int wave   = __builtin_amdgcn_readfirstlane(tid) >> 5;   // SGPR
    const int ptbase = blockIdx.x * NPTS;
    const int gpt    = ptbase + wave;                  // wave-uniform, == b*N+n
    const int b      = gpt >> 12;                      // /4096

    // ---- TDM #1 (wave 0 only): W f16 (16KB) -> wlds, plain 1-D tile --------
    if (tid < 32) {
        const unsigned long long wa = (unsigned long long)(uintptr_t)Wh;
        const unsigned int lw = (unsigned int)(uintptr_t)&wlds[0];
        u32x4 g0 = { 1u,                                // count=1
                     lw,                                // lds_addr
                     (unsigned int)wa,                  // global_addr[31:0]
                     (unsigned int)((wa >> 32) & 0x01FFFFFFu) | 0x80000000u }; // +type=2
        u32x8 g1 = { 0x00010000u,                       // data_size=1 (2B)
                     (unsigned int)(CC * TWOC) << 16,   // tensor_dim0[15:0]=8192
                     0u,                                // tensor_dim1=0 (1-D)
                     (unsigned int)(CC * TWOC) << 16,   // tile_dim0=8192
                     0u,                                // tile_dim1=0 (unused)
                     (unsigned int)(CC * TWOC),         // tensor_dim0_stride
                     0u, 0u };
        u32x4 gz = { 0u, 0u, 0u, 0u };
        tdm_load(g0, g1, gz, gz);
    }

    // ---- TDM #2 (each wave): gather 16 neighbor rows of x -> xrows[wave] ---
    {
        // 16 u16 row indices for this point (8 dwords), forced into SGPRs
        const unsigned int* ip =
            (const unsigned int*)(idx16 + (size_t)gpt * KNB);
        unsigned int i0 = __builtin_amdgcn_readfirstlane(ip[0]);
        unsigned int i1 = __builtin_amdgcn_readfirstlane(ip[1]);
        unsigned int i2 = __builtin_amdgcn_readfirstlane(ip[2]);
        unsigned int i3 = __builtin_amdgcn_readfirstlane(ip[3]);
        unsigned int i4 = __builtin_amdgcn_readfirstlane(ip[4]);
        unsigned int i5 = __builtin_amdgcn_readfirstlane(ip[5]);
        unsigned int i6 = __builtin_amdgcn_readfirstlane(ip[6]);
        unsigned int i7 = __builtin_amdgcn_readfirstlane(ip[7]);

        const unsigned long long xa =
            (unsigned long long)(uintptr_t)(x + (size_t)b * NN * CC);
        const unsigned int lx =
            (unsigned int)(uintptr_t)&xrows[wave * KNB * CC];
        u32x4 g0 = { 0x80000001u,                      // gather_mode | 16-bit idx | count=1
                     lx,                               // lds_addr
                     (unsigned int)xa,                 // global_addr[31:0]
                     (unsigned int)((xa >> 32) & 0x01FFFFFFu) | 0x80000000u }; // +type=2
        u32x8 g1 = { 0x00020000u,                      // data_size=2 (4B)
                     (unsigned int)CC << 16,           // tensor_dim0 = 64
                     (unsigned int)NN << 16,           // tensor_dim1 = 4096
                     (unsigned int)CC << 16,           // tile_dim0 = 64
                     (unsigned int)KNB,                // tile_dim1 = 16 indices
                     (unsigned int)CC,                 // tensor_dim0_stride = 64
                     0u, 0u };
        u32x4 g2 = { i0, i1, i2, i3 };                 // row indices 0..7
        u32x4 g3 = { i4, i5, i6, i7 };                 // row indices 8..15
        tdm_load(g0, g1, g2, g3);
    }

    __builtin_amdgcn_s_wait_tensorcnt(0);              // own gather (+W on wave0)
    __syncthreads();                                   // publish W to all waves

    // ---- per-wave edge build: edge[k][c] = {central, feat - central} f16 ---
    const int M  = lane & 15;                          // neighbor row / A row
    const int hl = lane >> 4;                          // half selector
    {
        const float* cen = x + (size_t)gpt * CC + hl * 32;
        const float* fr  = xrows + (wave * KNB + M) * CC + hl * 32;
        _Float16* er = edge + (wave * KNB + M) * ESTR + hl * 32;
#pragma unroll
        for (int c0 = 0; c0 < 32; c0 += 8) {
            const float4 c_0 = *(const float4*)(cen + c0);
            const float4 c_1 = *(const float4*)(cen + c0 + 4);
            const float4 f_0 = *(const float4*)(fr + c0);
            const float4 f_1 = *(const float4*)(fr + c0 + 4);
            h8 hc, hd;
            hc[0] = (_Float16)c_0.x; hd[0] = (_Float16)(f_0.x - c_0.x);
            hc[1] = (_Float16)c_0.y; hd[1] = (_Float16)(f_0.y - c_0.y);
            hc[2] = (_Float16)c_0.z; hd[2] = (_Float16)(f_0.z - c_0.z);
            hc[3] = (_Float16)c_0.w; hd[3] = (_Float16)(f_0.w - c_0.w);
            hc[4] = (_Float16)c_1.x; hd[4] = (_Float16)(f_1.x - c_1.x);
            hc[5] = (_Float16)c_1.y; hd[5] = (_Float16)(f_1.y - c_1.y);
            hc[6] = (_Float16)c_1.z; hd[6] = (_Float16)(f_1.z - c_1.z);
            hc[7] = (_Float16)c_1.w; hd[7] = (_Float16)(f_1.w - c_1.w);
            *(h8*)(er + c0)      = hc;                 // central half   [c]
            *(h8*)(er + CC + c0) = hd;                 // edge-diff half [64+c]
        }
    }
    // Same-wave LDS ops are in-order: this wave's edge writes are visible to
    // its own ds_loads below without a barrier (each wave reads only its tile).

    const _Float16* eptr = edge + wave * KNB * ESTR;

    // A fragments for the 4 K-steps (ISA 16-bit A layout:
    // VGPR0-3 -> K = 8*hl + 0..7 ; VGPR4-7 -> K = 16 + 8*hl + 0..7)
    v16h A[4];
#pragma unroll
    for (int s = 0; s < 4; ++s) {
        const _Float16* rowp = eptr + M * ESTR + s * 32 + hl * 8;
        const h8 lo = *(const h8*)rowp;
        const h8 hi = *(const h8*)(rowp + 16);
#pragma unroll
        for (int i = 0; i < 8; ++i) { A[s][i] = lo[i]; A[s][8 + i] = hi[i]; }
    }

#pragma unroll
    for (int t = 0; t < 4; ++t) {                      // output-channel tiles
        v8f acc = {};
        const int o = t * 16 + M;                      // this lane's channel
#pragma unroll
        for (int s = 0; s < 4; ++s) {                  // K steps of 32
            // B layout: lane = N column (o); K = 8*hl + 0..7, 16 + 8*hl + 0..7 (matches A).
            const _Float16* wp = wlds + o * TWOC + s * 32 + hl * 8;
            const h8 lo = *(const h8*)wp;
            const h8 hi = *(const h8*)(wp + 16);
            v16h Bf;
#pragma unroll
            for (int i = 0; i < 8; ++i) { Bf[i] = lo[i]; Bf[8 + i] = hi[i]; }
            acc = __builtin_amdgcn_wmma_f32_16x16x32_f16(
                false, A[s], false, Bf, (short)0, acc, false, false);
        }
        const float bo = bias[o];
        float s1 = 0.0f, s2 = 0.0f, mx = -INFINITY, mn = INFINITY;
#pragma unroll
        for (int r = 0; r < 8; ++r) {                  // k-rows r + 8*hl
            const float v = acc[r] + bo;
            s1 += v; s2 += v * v;
            mx = fmaxf(mx, v); mn = fminf(mn, v);
        }
        // combine the two M-halves (lane ^ 16 holds same channel o)
        s1 += __shfl_xor(s1, 16, 32);
        s2 += __shfl_xor(s2, 16, 32);
        mx  = fmaxf(mx, __shfl_xor(mx, 16, 32));
        mn  = fminf(mn, __shfl_xor(mn, 16, 32));
        if (lane < 16) {
            hmax[(size_t)gpt * CC + o] = mx;
            hmin[(size_t)gpt * CC + o] = mn;
            wsum[wave * CC + o]   = s1;                // unique slot per (wave,o)
            wsumsq[wave * CC + o] = s2;
        }
    }
    __syncthreads();

    // ---- deterministic per-block channel partials --------------------------
    if (tid < CC) {
        float S = 0.0f, S2 = 0.0f;
#pragma unroll
        for (int w = 0; w < NPTS; ++w) {
            S  += wsum[w * CC + tid];
            S2 += wsumsq[w * CC + tid];
        }
        partial[(size_t)blockIdx.x * 128 + tid]      = S;
        partial[(size_t)blockIdx.x * 128 + 64 + tid] = S2;
    }
}

// ---------------------------------------------------------------------------
// 4) Deterministic stats reduction: 8 fixed slices per channel-stat + fixed
//    combine order (bitwise reproducible across graph replays)
// ---------------------------------------------------------------------------
__global__ __launch_bounds__(1024) void k_reduce(const float* __restrict__ partial,
                                                 float* __restrict__ stats) {
    __shared__ float acc[1024];
    const int tid = threadIdx.x;
    const int c   = tid & 127;                         // channel-stat
    const int sl  = tid >> 7;                          // slice 0..7
    float s = 0.0f;
    for (int blk = sl; blk < 8192; blk += 8)
        s += partial[(size_t)blk * 128 + c];
    acc[tid] = s;
    __syncthreads();
    if (tid < 128) {
        float t = 0.0f;
#pragma unroll
        for (int k = 0; k < 8; ++k) t += acc[k * 128 + tid];
        stats[tid] = t;
    }
}

// ---------------------------------------------------------------------------
// 5) Finalize: BN affine (monotone) + ReLU + implicit max-over-k via hmax/hmin
// ---------------------------------------------------------------------------
__global__ void k_final(const float* __restrict__ stats,
                        const float* __restrict__ gamma,
                        const float* __restrict__ beta,
                        const float* __restrict__ hmax,
                        const float* __restrict__ hmin,
                        float* __restrict__ out, int total) {
    const int i = blockIdx.x * 256 + threadIdx.x;
    if (i >= total) return;
    const int o = i & 63;
    const float mean = stats[o] * CNT_INV;
    const float var  = stats[64 + o] * CNT_INV - mean * mean;
    const float sc   = gamma[o] * rsqrtf(var + 1e-5f);
    const float sh   = beta[o] - mean * sc;
    const float h    = (sc >= 0.0f) ? hmax[i] : hmin[i];
    out[i] = fmaxf(0.0f, sc * h + sh);
}

// ---------------------------------------------------------------------------
extern "C" void kernel_launch(void* const* d_in, const int* in_sizes, int n_in,
                              void* d_out, int out_size, void* d_ws, size_t ws_size,
                              hipStream_t stream) {
    const float* x     = (const float*)d_in[0];   // (8,4096,64)
    const float* xyz   = (const float*)d_in[1];   // (8,4096,3)
    const float* W     = (const float*)d_in[2];   // (64,128)
    const float* bias  = (const float*)d_in[3];   // (64)
    const float* gamma = (const float*)d_in[4];   // (64)
    const float* beta  = (const float*)d_in[5];   // (64)
    float* out = (float*)d_out;

    char* ws = (char*)d_ws;
    float*          stats = (float*)(ws + WS_SUM);
    _Float16*       Wh    = (_Float16*)(ws + WS_WH);
    unsigned short* idx16 = (unsigned short*)(ws + WS_IDX);
    float*          hmax  = (float*)(ws + WS_HMAX);
    float*          hmin  = (float*)(ws + WS_HMIN);
    float*          part  = (float*)(ws + WS_PART);

    k_wconv<<<(CC * TWOC + 255) / 256, 256, 0, stream>>>(W, Wh);
    k_knn<<<BB * (NN / 256), 256, 0, stream>>>(xyz, idx16);
    k_edgeconv<<<(BB * NN) / NPTS, 128, 0, stream>>>(x, bias, Wh, idx16,
                                                     hmax, hmin, part);
    k_reduce<<<1, 1024, 0, stream>>>(part, stats);
    k_final<<<(out_size + 255) / 256, 256, 0, stream>>>(stats, gamma, beta,
                                                        hmax, hmin, out, out_size);
}
